// AttnConvLayer_3135326126344
// MI455X (gfx1250) — compile-verified, matrix operand order checked by reading, not run on hardware
//
#include <hip/hip_runtime.h>
#include <hip/hip_bf16.h>

// ---------------------------------------------------------------------------
// AttnConvLayer for MI455X (gfx1250, wave32, WMMA).
//
// Algebraic rewrite: h_ss = cat(s_feat[src], efeat_ss) @ Wss + b
//   = (s_feat @ Wss[:64])[src] + efeat_ss @ Wss[64:74] + b
// so per-edge GEMMs collapse to per-node GEMMs + tiny per-edge matvecs, and
// the attention logit's dst half is precomputed per node. The workload is
// then HBM/scatter bound; dense math runs on v_wmma_f32_16x16x32_f16.
//
// Round-1 codegen fixes:
//  * W staged TRANSPOSED in LDS so B fragments load as 2x ds_load_b128
//    (round 0 emitted 16x ds_load_u16 + v_mov_b16 merges per fragment).
//  * D tile staged through LDS (union-aliased) -> coalesced float4 stores
//    instead of 32 scattered b32 stores per lane.
// ---------------------------------------------------------------------------

typedef __attribute__((ext_vector_type(16))) _Float16 v16h;
typedef __attribute__((ext_vector_type(8)))  float    v8f;

#define N_S_  100000
#define N_O_  100000
#define E_SS_ 1000000
#define E_OS_ 1000000
#define E_F_  1000000
#define E_B_  1000000

// ---------------------------------------------------------------------------
// WMMA GEMM: Y[N,64] = act(X[N,64]) @ W[64,64] (+bias) (+= existing Y)
// Block = 128 threads = 4 waves; block computes a 64x64 output tile; each
// wave owns 16 rows x all four 16x16 col tiles; K=64 -> 8 WMMAs per wave.
// Fragment layouts per ISA 7.12.2:
//   A: lane L -> row (L&15); K-half sel = (L>>4); elems 0-7 = K kb..kb+7,
//      elems 8-15 = K kb+16..kb+23 (kb = 32*kstep + 8*(L>>4)).
//   B: mirrored (lane holds col), same K element pattern -> read from Wt[n][k].
//   C/D: VGPR r -> M = r + 8*(L>>4), N = (L&15).
// Row pitch 72 halfs keeps every 8-half run 16B-aligned for ds_load_b128.
// ---------------------------------------------------------------------------
template<bool RELU_IN, bool ACCUM, bool BIAS>
__global__ __launch_bounds__(128)
void gemm64_wmma(const float* __restrict__ X, const float* __restrict__ W,
                 const float* __restrict__ bias, float* __restrict__ Y, int nrows)
{
    __shared__ __align__(16) union ShTile {
        struct { _Float16 X[64][72]; _Float16 Wt[64][72]; } in;  // 18 KB
        float out[64][64];                                       // 16 KB
    } sh;
    __shared__ float sB[64];

    const int tid  = threadIdx.x;
    const int row0 = blockIdx.x * 64;

    // Stage W transposed: Wt[n][k] = W[k][n]  (scattered LDS writes, once)
    for (int i = tid; i < 64 * 64; i += 128) {
        const int k = i >> 6, n = i & 63;
        sh.in.Wt[n][k] = (_Float16)W[i];
    }
    if (BIAS) { if (tid < 64) sB[tid] = bias[tid]; }
    // Stage X tile (coalesced global loads, f32 -> f16)
    for (int i = tid; i < 64 * 64; i += 128) {
        const int r = i >> 6, c = i & 63;
        const int gr = row0 + r;
        float v = (gr < nrows) ? X[gr * 64 + c] : 0.0f;
        if (RELU_IN) v = v > 0.0f ? v : 0.0f;
        sh.in.X[r][c] = (_Float16)v;
    }
    __syncthreads();

    const int wave = tid >> 5;          // wave32: 4 waves / 128 threads
    const int lane = tid & 31;
    const int hsel = lane >> 4;
    const int l15  = lane & 15;
    const int rowL = wave * 16 + l15;

    v8f acc[4];
#pragma unroll
    for (int ct = 0; ct < 4; ++ct) acc[ct] = (v8f){0, 0, 0, 0, 0, 0, 0, 0};

#pragma unroll
    for (int ks = 0; ks < 2; ++ks) {
        const int kb = ks * 32 + hsel * 8;
        v16h a;
#pragma unroll
        for (int e = 0; e < 8; ++e) a[e]     = sh.in.X[rowL][kb + e];
#pragma unroll
        for (int e = 0; e < 8; ++e) a[8 + e] = sh.in.X[rowL][kb + 16 + e];
#pragma unroll
        for (int ct = 0; ct < 4; ++ct) {
            const int col = ct * 16 + l15;
            v16h b;
#pragma unroll
            for (int e = 0; e < 8; ++e) b[e]     = sh.in.Wt[col][kb + e];
#pragma unroll
            for (int e = 0; e < 8; ++e) b[8 + e] = sh.in.Wt[col][kb + 16 + e];
            acc[ct] = __builtin_amdgcn_wmma_f32_16x16x32_f16(
                false, a, false, b, (short)0, acc[ct], false, false);
        }
    }

    // Stage D tile in LDS, then coalesced float4 epilogue.
    __syncthreads();
#pragma unroll
    for (int ct = 0; ct < 4; ++ct)
#pragma unroll
        for (int r = 0; r < 8; ++r)
            sh.out[wave * 16 + r + 8 * hsel][ct * 16 + l15] = acc[ct][r];
    __syncthreads();

    for (int i = tid; i < 64 * 16; i += 128) {      // 64 rows x 16 float4
        const int r = i >> 4, c4 = i & 15;
        const int grow = row0 + r;
        if (grow < nrows) {
            float4 v = ((const float4*)sh.out[r])[c4];
            if (BIAS) {
                v.x += sB[c4 * 4 + 0]; v.y += sB[c4 * 4 + 1];
                v.z += sB[c4 * 4 + 2]; v.w += sB[c4 * 4 + 3];
            }
            float4* yp = (float4*)(Y + (size_t)grow * 64) + c4;
            if (ACCUM) {
                const float4 o = *yp;
                v.x += o.x; v.y += o.y; v.z += o.z; v.w += o.w;
            }
            *yp = v;
        }
    }
}

// ---------------------------------------------------------------------------
// a_s[n] = h_s[n] . attn_w[64:128]   (wave-per-node, shuffle reduction)
// ---------------------------------------------------------------------------
__global__ __launch_bounds__(256)
void attn_dst_dot(const float* __restrict__ h_s, const float* __restrict__ attn_w,
                  float* __restrict__ a_s, int n)
{
    const int lane = threadIdx.x & 31;
    const int node = blockIdx.x * 8 + (threadIdx.x >> 5);
    if (node >= n) return;
    float p = h_s[node * 64 + lane]      * attn_w[64 + lane]
            + h_s[node * 64 + 32 + lane] * attn_w[96 + lane];
#pragma unroll
    for (int off = 16; off > 0; off >>= 1) p += __shfl_xor(p, off, 32);
    if (lane == 0) a_s[node] = p;
}

// ---------------------------------------------------------------------------
// Edge pass 1: logits -> exp -> denominator scatter. Wave-per-edge; lane l
// owns channels l and l+32 (coalesced 64-float row gather).
// ---------------------------------------------------------------------------
template<int EF>
__global__ __launch_bounds__(256)
void edge_pass1(const int* __restrict__ src, const int* __restrict__ dst,
                const float* __restrict__ g, const float* __restrict__ efeat,
                const float* __restrict__ We, const float* __restrict__ be,
                const float* __restrict__ attn_w, const float* __restrict__ attn_b,
                const float* __restrict__ a_dst,
                float* __restrict__ nom, float* __restrict__ den, int nedges)
{
    __shared__ float sWe[EF][64];
    __shared__ float sA[64];
    __shared__ float sBe[64];
    const int tid = threadIdx.x;
    for (int i = tid; i < EF * 64; i += 256) sWe[i / 64][i % 64] = We[i];
    if (tid < 64) { sA[tid] = attn_w[tid]; sBe[tid] = be[tid]; }
    __syncthreads();

    const int lane = tid & 31;
    const int e = blockIdx.x * 8 + (tid >> 5);
    if (e >= nedges) return;
    const int s = src[e], d = dst[e];
    float h0 = g[s * 64 + lane]      + sBe[lane];
    float h1 = g[s * 64 + 32 + lane] + sBe[32 + lane];
#pragma unroll
    for (int j = 0; j < EF; ++j) {
        const float f = efeat[e * EF + j];
        h0 += f * sWe[j][lane];
        h1 += f * sWe[j][32 + lane];
    }
    float part = h0 * sA[lane] + h1 * sA[32 + lane];
#pragma unroll
    for (int off = 16; off > 0; off >>= 1) part += __shfl_xor(part, off, 32);
    if (lane == 0) {
        float logit = part + a_dst[d] + attn_b[0];
        logit = logit > 0.0f ? logit : 0.01f * logit;   // leaky_relu(0.01)
        const float nm = __expf(logit);
        nom[e] = nm;
        atomicAdd(&den[d], nm);
    }
}

// ---------------------------------------------------------------------------
// Edge pass 2: recompute h (cheaper than spilling 256 MB of h_e), scale by
// alpha = nom/den[dst], scatter into z with f32 atomics.
// ---------------------------------------------------------------------------
template<int EF>
__global__ __launch_bounds__(256)
void edge_pass2(const int* __restrict__ src, const int* __restrict__ dst,
                const float* __restrict__ g, const float* __restrict__ efeat,
                const float* __restrict__ We, const float* __restrict__ be,
                const float* __restrict__ nom, const float* __restrict__ den,
                float* __restrict__ z, int nedges)
{
    __shared__ float sWe[EF][64];
    __shared__ float sBe[64];
    const int tid = threadIdx.x;
    for (int i = tid; i < EF * 64; i += 256) sWe[i / 64][i % 64] = We[i];
    if (tid < 64) sBe[tid] = be[tid];
    __syncthreads();

    const int lane = tid & 31;
    const int e = blockIdx.x * 8 + (tid >> 5);
    if (e >= nedges) return;
    const int s = src[e], d = dst[e];
    float h0 = g[s * 64 + lane]      + sBe[lane];
    float h1 = g[s * 64 + 32 + lane] + sBe[32 + lane];
#pragma unroll
    for (int j = 0; j < EF; ++j) {
        const float f = efeat[e * EF + j];
        h0 += f * sWe[j][lane];
        h1 += f * sWe[j][32 + lane];
    }
    const float alpha = nom[e] / den[d];
    atomicAdd(&z[d * 64 + lane],      alpha * h0);
    atomicAdd(&z[d * 64 + 32 + lane], alpha * h1);
}

// ---------------------------------------------------------------------------
// out[dst] += feat[src] (copy_u + segment_sum), wave-per-edge.
// ---------------------------------------------------------------------------
__global__ __launch_bounds__(256)
void scatter_sum64(const int* __restrict__ src, const int* __restrict__ dst,
                   const float* __restrict__ feat, float* __restrict__ out, int nedges)
{
    const int lane = threadIdx.x & 31;
    const int e = blockIdx.x * 8 + (threadIdx.x >> 5);
    if (e >= nedges) return;
    const int s = src[e], d = dst[e];
    atomicAdd(&out[d * 64 + lane],      feat[s * 64 + lane]);
    atomicAdd(&out[d * 64 + 32 + lane], feat[s * 64 + 32 + lane]);
}

__global__ void zero_f32(float* __restrict__ p, size_t n)
{
    const size_t i = (size_t)blockIdx.x * blockDim.x + threadIdx.x;
    if (i < n) p[i] = 0.0f;
}

// ---------------------------------------------------------------------------
extern "C" void kernel_launch(void* const* d_in, const int* in_sizes, int n_in,
                              void* d_out, int out_size, void* d_ws, size_t ws_size,
                              hipStream_t stream)
{
    (void)in_sizes; (void)n_in; (void)out_size; (void)ws_size;

    const float* s_feat   = (const float*)d_in[0];
    const float* o_feat   = (const float*)d_in[1];
    const float* efeat_ss = (const float*)d_in[2];
    const float* efeat_os = (const float*)d_in[3];
    const int*   ss_src   = (const int*)d_in[4];
    const int*   ss_dst   = (const int*)d_in[5];
    const int*   os_src   = (const int*)d_in[6];
    const int*   os_dst   = (const int*)d_in[7];
    const int*   fwd_src  = (const int*)d_in[8];
    const int*   fwd_dst  = (const int*)d_in[9];
    const int*   bwd_src  = (const int*)d_in[10];
    const int*   bwd_dst  = (const int*)d_in[11];
    const float* Ws_w   = (const float*)d_in[12];
    const float* Ws_b   = (const float*)d_in[13];
    const float* Wss_w  = (const float*)d_in[14];
    const float* Wss_b  = (const float*)d_in[15];
    const float* Wos_w  = (const float*)d_in[16];
    const float* Wos_b  = (const float*)d_in[17];
    const float* attn_w = (const float*)d_in[18];
    const float* attn_b = (const float*)d_in[19];
    const float* Win_w   = (const float*)d_in[20];
    const float* Win_b   = (const float*)d_in[21];
    const float* Wself_w = (const float*)d_in[22];
    const float* Wself_b = (const float*)d_in[23];
    const float* Wout_w  = (const float*)d_in[24];
    const float* Wout_b  = (const float*)d_in[25];
    const float* Wo_w    = (const float*)d_in[26];
    const float* Wo_b    = (const float*)d_in[27];

    // ---- workspace carve-up (~214 MB of f32) ----
    float* ws = (float*)d_ws;
    size_t off = 0;
    auto alloc = [&](size_t n) { float* p = ws + off; off += n; return p; };
    float* g_ss      = alloc((size_t)N_S_ * 64);  // s_feat @ Wss[:64]
    float* h_s       = alloc((size_t)N_S_ * 64);  // s_feat @ Ws + b
    float* g_os      = alloc((size_t)N_O_ * 64);  // o_feat @ Wos[:64]
    float* h_in_all  = alloc((size_t)N_O_ * 64);
    float* h_self    = alloc((size_t)N_O_ * 64);
    float* h_out_all = alloc((size_t)N_O_ * 64);
    float* h_in_agg  = alloc((size_t)N_O_ * 64);
    float* h_out_agg = alloc((size_t)N_O_ * 64);
    float* a_s       = alloc(N_S_);
    float* den_ss    = alloc(N_S_);
    float* den_os    = alloc(N_S_);
    float* nom_ss    = alloc(E_SS_);
    float* nom_os    = alloc(E_OS_);

    float* z = (float*)d_out;                       // [N_S, 64]
    float* x = (float*)d_out + (size_t)N_S_ * 64;   // [N_O, 64]

    auto zero = [&](float* p, size_t n) {
        zero_f32<<<(unsigned)((n + 255) / 256), 256, 0, stream>>>(p, n);
    };
    zero(z, (size_t)N_S_ * 64);
    zero(den_ss, N_S_);
    zero(den_os, N_S_);
    zero(h_in_agg, (size_t)N_O_ * 64);
    zero(h_out_agg, (size_t)N_O_ * 64);

    const int gS = (N_S_ + 63) / 64, gO = (N_O_ + 63) / 64;

    // ---- per-node WMMA GEMMs ----
    gemm64_wmma<false, false, false><<<gS, 128, 0, stream>>>(s_feat, Wss_w,  nullptr, g_ss,      N_S_);
    gemm64_wmma<false, false, true ><<<gS, 128, 0, stream>>>(s_feat, Ws_w,   Ws_b,    h_s,       N_S_);
    gemm64_wmma<false, false, false><<<gO, 128, 0, stream>>>(o_feat, Wos_w,  nullptr, g_os,      N_O_);
    gemm64_wmma<false, false, true ><<<gO, 128, 0, stream>>>(o_feat, Win_w,  Win_b,   h_in_all,  N_O_);
    gemm64_wmma<false, false, true ><<<gO, 128, 0, stream>>>(o_feat, Wself_w,Wself_b, h_self,    N_O_);
    gemm64_wmma<false, false, true ><<<gO, 128, 0, stream>>>(o_feat, Wout_w, Wout_b,  h_out_all, N_O_);

    attn_dst_dot<<<(N_S_ + 7) / 8, 256, 0, stream>>>(h_s, attn_w, a_s, N_S_);

    // ---- attention edge passes (edge-feature rows of Wss/Wos start at row 64) ----
    edge_pass1<10><<<(E_SS_ + 7) / 8, 256, 0, stream>>>(ss_src, ss_dst, g_ss, efeat_ss,
        Wss_w + 64 * 64, Wss_b, attn_w, attn_b, a_s, nom_ss, den_ss, E_SS_);
    edge_pass1<2 ><<<(E_OS_ + 7) / 8, 256, 0, stream>>>(os_src, os_dst, g_os, efeat_os,
        Wos_w + 64 * 64, Wos_b, attn_w, attn_b, a_s, nom_os, den_os, E_OS_);
    edge_pass2<10><<<(E_SS_ + 7) / 8, 256, 0, stream>>>(ss_src, ss_dst, g_ss, efeat_ss,
        Wss_w + 64 * 64, Wss_b, nom_ss, den_ss, z, E_SS_);
    edge_pass2<2 ><<<(E_OS_ + 7) / 8, 256, 0, stream>>>(os_src, os_dst, g_os, efeat_os,
        Wos_w + 64 * 64, Wos_b, nom_os, den_os, z, E_OS_);

    // ---- fwd/bwd copy_u + segment_sum ----
    scatter_sum64<<<(E_F_ + 7) / 8, 256, 0, stream>>>(fwd_src, fwd_dst, h_in_all,  h_in_agg,  E_F_);
    scatter_sum64<<<(E_B_ + 7) / 8, 256, 0, stream>>>(bwd_src, bwd_dst, h_out_all, h_out_agg, E_B_);

    // ---- x = relu(cat[h_in, h_self, h_out]) @ Wo + b, decomposed over K ----
    gemm64_wmma<true, false, false><<<gO, 128, 0, stream>>>(h_in_agg,  Wo_w + 0 * 64 * 64, nullptr, x, N_O_);
    gemm64_wmma<true, true,  false><<<gO, 128, 0, stream>>>(h_self,    Wo_w + 1 * 64 * 64, nullptr, x, N_O_);
    gemm64_wmma<true, true,  true ><<<gO, 128, 0, stream>>>(h_out_agg, Wo_w + 2 * 64 * 64, Wo_b,    x, N_O_);
}
